// EGNNLayer_7885559956063
// MI455X (gfx1250) — compile-verified, matrix operand order checked by reading, not run on hardware
//
#include <hip/hip_runtime.h>

typedef __attribute__((ext_vector_type(16))) _Float16 v16h;
typedef __attribute__((ext_vector_type(8)))  _Float16 v8h;   // 16 B
typedef __attribute__((ext_vector_type(8)))  float    v8f;
typedef __attribute__((ext_vector_type(4)))  int      v4i;   // 16 B

#define HD 64
#define KP_MSG 160   // 2H+EA+1 = 131 padded to 5*32
#define KP_NF  128
#define WPB_E 2      // waves per block, edge kernel (32 edges per wave)
#define WPB_N 4      // waves per block, node kernel (16 nodes per wave)

#define AS1 __attribute__((address_space(1)))
#define AS3 __attribute__((address_space(3)))

#if __has_builtin(__builtin_amdgcn_global_load_async_to_lds_b128)
#define HAVE_ASYNC_LDS 1
#endif

// Copy one 64-element f16 row (128 B) global -> LDS.
__device__ __forceinline__ void copy_row64h(const _Float16* g, _Float16* l) {
#ifdef HAVE_ASYNC_LDS
  AS1 v4i* gp = (AS1 v4i*)g;
  AS3 v4i* lp = (AS3 v4i*)l;
#pragma unroll
  for (int q = 0; q < 8; ++q)
    __builtin_amdgcn_global_load_async_to_lds_b128(gp + q, lp + q, 0, 0);
#else
  const v8h* s = (const v8h*)g;
  v8h* d = (v8h*)l;
#pragma unroll
  for (int q = 0; q < 8; ++q) d[q] = s[q];
#endif
}

__device__ __forceinline__ void async_wait0() {
#ifdef HAVE_ASYNC_LDS
#if __has_builtin(__builtin_amdgcn_s_wait_asynccnt)
  __builtin_amdgcn_s_wait_asynccnt(0);
#else
  asm volatile("s_wait_asynccnt 0x0" ::: "memory");
#endif
#endif
}

// Intra-wave LDS ordering: compiler barrier + CDNA5 split DS counter wait.
#define LDS_FENCE() asm volatile("s_wait_dscnt 0x0" ::: "memory")

// ---------------- WMMA fragment helpers (ISA 7.12.2 16-bit layouts) ----------

__device__ __forceinline__ int frag_k(int lane, int p) {
  int kb = (lane & 16) ? 8 : 0;
  return kb + ((p < 4) ? (2 * p) : (16 + 2 * (p - 4)));
}

// A: 16 rows x (32*nc) cols, row-major f16 in LDS. Per lane this is two
// contiguous 16B runs (k 0..7 and 16..23, +8 for hi lanes) -> 2x ds_load_b128.
__device__ __forceinline__ v16h load_A(const _Float16* src, int stride, int kc) {
  int lane = threadIdx.x & 31;
  const _Float16* base = src + (lane & 15) * stride + kc * 32;
  v16h a;
#pragma unroll
  for (int p = 0; p < 8; ++p) {
    int k = frag_k(lane, p);                      // always even -> 4B aligned
    unsigned w = *(const unsigned*)(base + k);
    a[2 * p]     = __builtin_bit_cast(_Float16, (unsigned short)(w & 0xffffu));
    a[2 * p + 1] = __builtin_bit_cast(_Float16, (unsigned short)(w >> 16));
  }
  return a;
}

// B: pre-packed fragment-linear; one contiguous 32B load per lane.
__device__ __forceinline__ v16h load_B(const _Float16* packed, int nt, int kc, int nc) {
  int lane = threadIdx.x & 31;
  return ((const v16h*)packed)[(nt * nc + kc) * 32 + lane];
}

__device__ __forceinline__ float silu_f(float x) { return x / (1.f + __expf(-x)); }

// Two M-tiles (32 rows) x 64 cols: B fragment loaded once, used by 2 WMMAs.
__device__ __forceinline__ void gemm64_m2(const _Float16* ldsA, int strideA, int nc,
                                          const _Float16* packedB, const float* bias,
                                          v8f out[2][4]) {
  int lane = threadIdx.x & 31;
#pragma unroll
  for (int nt = 0; nt < 4; ++nt) {
    v8f c0 = {}, c1 = {};
    for (int kc = 0; kc < nc; ++kc) {
      v16h b  = load_B(packedB, nt, kc, nc);
      v16h a0 = load_A(ldsA, strideA, kc);
      v16h a1 = load_A(ldsA + 16 * strideA, strideA, kc);
      c0 = __builtin_amdgcn_wmma_f32_16x16x32_f16(false, a0, false, b,
                                                  (short)0, c0, false, false);
      c1 = __builtin_amdgcn_wmma_f32_16x16x32_f16(false, a1, false, b,
                                                  (short)0, c1, false, false);
    }
    float bv = bias[nt * 16 + (lane & 15)];
#pragma unroll
    for (int r = 0; r < 8; ++r) { c0[r] += bv; c1[r] += bv; }
    out[0][nt] = c0; out[1][nt] = c1;
  }
}

// Single M-tile version (node kernel).
__device__ __forceinline__ void gemm64(const _Float16* ldsA, int strideA, int nc,
                                       const _Float16* packedB, const float* bias,
                                       v8f out[4]) {
  int lane = threadIdx.x & 31;
#pragma unroll
  for (int nt = 0; nt < 4; ++nt) {
    v8f c = {};
    for (int kc = 0; kc < nc; ++kc) {
      v16h a = load_A(ldsA, strideA, kc);
      v16h b = load_B(packedB, nt, kc, nc);
      c = __builtin_amdgcn_wmma_f32_16x16x32_f16(false, a, false, b,
                                                 (short)0, c, false, false);
    }
    float bv = bias[nt * 16 + (lane & 15)];
#pragma unroll
    for (int r = 0; r < 8; ++r) c[r] += bv;
    out[nt] = c;
  }
}

__device__ __forceinline__ void store_tiles(_Float16* dst, int stride,
                                            const v8f t[4], bool act) {
  int lane = threadIdx.x & 31;
  int rb = (lane & 16) ? 8 : 0;
  int n0 = lane & 15;
#pragma unroll
  for (int nt = 0; nt < 4; ++nt)
#pragma unroll
    for (int r = 0; r < 8; ++r) {
      float v = t[nt][r];
      if (act) v = silu_f(v);
      dst[(rb + r) * stride + nt * 16 + n0] = (_Float16)v;
    }
}

__device__ __forceinline__ void store_tiles_m2(_Float16* dst, int stride,
                                               const v8f t[2][4], bool act) {
  store_tiles(dst, stride, t[0], act);
  store_tiles(dst + 16 * stride, stride, t[1], act);
}

// ---------------- pre-pass kernels -------------------------------------------

__global__ void pack_B_kernel(const float* __restrict__ W, _Float16* __restrict__ out,
                              int Kact, int KPv) {
  int i = blockIdx.x * blockDim.x + threadIdx.x;
  int total = 64 * KPv;                 // 4 ntiles * (KPv/32) kchunks * 32 lanes * 16
  if (i >= total) return;
  int j    = i & 15;
  int lane = (i >> 4) & 31;
  int frag = i >> 9;
  int nc = KPv >> 5;
  int nt = frag / nc, kc = frag % nc;
  int n = nt * 16 + (lane & 15);
  int p = j >> 1;
  int k = kc * 32 + ((lane & 16) ? 8 : 0)
        + ((p < 4) ? 2 * p : 16 + 2 * (p - 4)) + (j & 1);
  out[i] = (_Float16)((k < Kact) ? W[k * 64 + n] : 0.f);
}

__global__ void cvt_feat_kernel(const float* __restrict__ f, _Float16* __restrict__ o,
                                int n) {
  int i = blockIdx.x * blockDim.x + threadIdx.x;
  if (i < n) o[i] = (_Float16)f[i];
}

__global__ void zero_kernel(float* __restrict__ p, int n) {
  int i = blockIdx.x * blockDim.x + threadIdx.x;
  if (i < n) p[i] = 0.f;
}

// ---------------- edge kernel: msg MLP + pos MLP + scatter -------------------
// One wave = 32 edges = two 16-row M-tiles sharing every B fragment.

__global__ __launch_bounds__(32 * WPB_E)
void egnn_edge_kernel(const _Float16* __restrict__ feat16,
                      const float* __restrict__ node_pos,
                      const int* __restrict__ erow, const int* __restrict__ ecol,
                      const float* __restrict__ eattr,
                      const _Float16* __restrict__ w1p, const float* __restrict__ b1,
                      const _Float16* __restrict__ w2p, const float* __restrict__ b2,
                      const _Float16* __restrict__ pw1p, const float* __restrict__ pb1,
                      const float* __restrict__ pw2, const float* __restrict__ pb2,
                      float* __restrict__ msg_agg, float* __restrict__ pos_agg,
                      float* __restrict__ cnt, int Eedges) {
  __shared__ _Float16 s_in [WPB_E][32 * KP_MSG];
  __shared__ _Float16 s_h  [WPB_E][32 * HD];
  __shared__ _Float16 s_msg[WPB_E][32 * HD];
  __shared__ float    s_diff[WPB_E][32][3];
  __shared__ int      s_node[WPB_E][32];

  int wave = threadIdx.x >> 5;
  int lane = threadIdx.x & 31;
  int tile = blockIdx.x * WPB_E + wave;
  int e0 = tile * 32;
  if (e0 >= Eedges) return;              // per-wave tiles, no block barriers

  _Float16* in     = s_in[wave];
  _Float16* hbuf   = s_h[wave];
  _Float16* msgbuf = s_msg[wave];

  // ---- stage 32 edges: one lane per edge; f16 rows via async global->LDS
  {
    int e = lane;
    int eg = e0 + e; if (eg >= Eedges) eg = Eedges - 1;
    int r = erow[eg], c = ecol[eg];
    _Float16* row = in + e * KP_MSG;
    copy_row64h(feat16 + (size_t)r * 64, row);        // feat[row]  (async)
    copy_row64h(feat16 + (size_t)c * 64, row + 64);   // feat[col]  (async)
    float dx = node_pos[r * 3 + 0] - node_pos[c * 3 + 0];
    float dy = node_pos[r * 3 + 1] - node_pos[c * 3 + 1];
    float dz = node_pos[r * 3 + 2] - node_pos[c * 3 + 2];
    s_diff[wave][e][0] = dx; s_diff[wave][e][1] = dy; s_diff[wave][e][2] = dz;
    s_node[wave][e] = r;
    row[128] = (_Float16)eattr[eg * 2 + 0];
    row[129] = (_Float16)eattr[eg * 2 + 1];
    row[130] = (_Float16)(dx * dx + dy * dy + dz * dz);
#pragma unroll
    for (int k = 131; k < KP_MSG; ++k) row[k] = (_Float16)0.f;   // pad
  }
  async_wait0();
  LDS_FENCE();

  v8f t[2][4];
  // msg layer 1: [32,160] x [160,64], silu
  gemm64_m2(in, KP_MSG, KP_MSG / 32, w1p, b1, t);
  store_tiles_m2(hbuf, HD, t, true);
  LDS_FENCE();
  // msg layer 2: [32,64] x [64,64], silu -> msg ; scatter-add to msg_agg
  gemm64_m2(hbuf, HD, 2, w2p, b2, t);
  {
    int rb = (lane & 16) ? 8 : 0;
    int n0 = lane & 15;
#pragma unroll
    for (int mt = 0; mt < 2; ++mt)
#pragma unroll
      for (int nt = 0; nt < 4; ++nt)
#pragma unroll
        for (int r = 0; r < 8; ++r) {
          float v = silu_f(t[mt][nt][r]);
          int m = mt * 16 + rb + r;
          msgbuf[m * HD + nt * 16 + n0] = (_Float16)v;
          if (e0 + m < Eedges)
            atomicAdd(&msg_agg[s_node[wave][m] * HD + nt * 16 + n0], v);
        }
  }
  LDS_FENCE();
  if (e0 + lane < Eedges)
    atomicAdd(&cnt[s_node[wave][lane]], 1.f);

  // pos MLP layer 1: [32,64] x [64,64], silu
  gemm64_m2(msgbuf, HD, 2, pw1p, pb1, t);
  store_tiles_m2(hbuf, HD, t, true);
  LDS_FENCE();
  // pos head: rank-1 (64 -> 1) dot per edge (full wave), scale diff, scatter
  {
    int e = lane;
    if (e0 + e < Eedges) {
      float s = pb2[0];
      for (int k = 0; k < 64; ++k) s += (float)hbuf[e * HD + k] * pw2[k];
      int node = s_node[wave][e];
      atomicAdd(&pos_agg[node * 3 + 0], s_diff[wave][e][0] * s);
      atomicAdd(&pos_agg[node * 3 + 1], s_diff[wave][e][1] * s);
      atomicAdd(&pos_agg[node * 3 + 2], s_diff[wave][e][2] * s);
    }
  }
}

// ---------------- node kernel: feat MLP + vel MLP + position update ----------

__global__ __launch_bounds__(32 * WPB_N)
void egnn_node_kernel(const _Float16* __restrict__ feat16,
                      const float* __restrict__ node_pos,
                      const float* __restrict__ node_vel,
                      const _Float16* __restrict__ nfw1p, const float* __restrict__ nfb1,
                      const _Float16* __restrict__ nfw2p, const float* __restrict__ nfb2,
                      const _Float16* __restrict__ vw1p,  const float* __restrict__ vb1,
                      const float* __restrict__ vw2, const float* __restrict__ vb2,
                      const float* __restrict__ msg_agg, const float* __restrict__ pos_agg,
                      const float* __restrict__ cnt,
                      float* __restrict__ out_feat, float* __restrict__ out_pos, int Nn) {
  __shared__ _Float16 s_in[WPB_N][16 * KP_NF];
  __shared__ _Float16 s_h [WPB_N][16 * HD];

  int wave = threadIdx.x >> 5;
  int lane = threadIdx.x & 31;
  int tile = blockIdx.x * WPB_N + wave;
  int n0 = tile * 16;
  if (n0 >= Nn) return;

  _Float16* in   = s_in[wave];
  _Float16* hbuf = s_h[wave];

  // stage [feat(64) | msg_mean(64)] per node
  {
    int e = lane & 15;
    int half = lane >> 4;
    int node = n0 + e; if (node >= Nn) node = Nn - 1;
    _Float16* row = in + e * KP_NF + half * 64;
    if (half == 0) {
      copy_row64h(feat16 + (size_t)node * 64, row);   // async f16 row
    } else {
      float inv = 1.f / fmaxf(cnt[node], 1.f);
#pragma unroll 4
      for (int k = 0; k < 64; ++k) row[k] = (_Float16)(msg_agg[node * 64 + k] * inv);
    }
  }
  async_wait0();
  LDS_FENCE();

  v8f t[4];
  // feat MLP layer 1: [16,128] x [128,64], silu
  gemm64(in, KP_NF, 4, nfw1p, nfb1, t);
  store_tiles(hbuf, HD, t, true);
  LDS_FENCE();
  // feat MLP layer 2: no activation, write new_feat
  gemm64(hbuf, HD, 2, nfw2p, nfb2, t);
  {
    int rb = (lane & 16) ? 8 : 0;
    int nn = lane & 15;
#pragma unroll
    for (int nt = 0; nt < 4; ++nt)
#pragma unroll
      for (int r = 0; r < 8; ++r) {
        int node = n0 + rb + r;
        if (node < Nn) out_feat[node * 64 + nt * 16 + nn] = t[nt][r];
      }
  }
  LDS_FENCE();
  // vel MLP layer 1 over feat half of s_in (stride KP_NF, 2 K-chunks), silu
  gemm64(in, KP_NF, 2, vw1p, vb1, t);
  store_tiles(hbuf, HD, t, true);
  LDS_FENCE();
  // vel head + position update
  if (lane < 16) {
    int node = n0 + lane;
    if (node < Nn) {
      float inv = 1.f / fmaxf(cnt[node], 1.f);
      float s = vb2[0];
      for (int k = 0; k < 64; ++k) s += (float)hbuf[lane * HD + k] * vw2[k];
      for (int d = 0; d < 3; ++d)
        out_pos[node * 3 + d] = node_pos[node * 3 + d]
                              + pos_agg[node * 3 + d] * inv
                              + s * node_vel[node * 3 + d];
    }
  }
}

// ---------------- host launch ------------------------------------------------

extern "C" void kernel_launch(void* const* d_in, const int* in_sizes, int n_in,
                              void* d_out, int out_size, void* d_ws, size_t ws_size,
                              hipStream_t stream) {
  const float* node_feat = (const float*)d_in[0];
  const float* node_pos  = (const float*)d_in[1];
  const float* node_vel  = (const float*)d_in[2];
  const int*   edge_row  = (const int*)d_in[3];
  const int*   edge_col  = (const int*)d_in[4];
  const float* edge_attr = (const float*)d_in[5];
  const float* msg_w1 = (const float*)d_in[6],  *msg_b1 = (const float*)d_in[7];
  const float* msg_w2 = (const float*)d_in[8],  *msg_b2 = (const float*)d_in[9];
  const float* pos_w1 = (const float*)d_in[10], *pos_b1 = (const float*)d_in[11];
  const float* pos_w2 = (const float*)d_in[12], *pos_b2 = (const float*)d_in[13];
  const float* nf_w1  = (const float*)d_in[14], *nf_b1  = (const float*)d_in[15];
  const float* nf_w2  = (const float*)d_in[16], *nf_b2  = (const float*)d_in[17];
  const float* vel_w1 = (const float*)d_in[18], *vel_b1 = (const float*)d_in[19];
  const float* vel_w2 = (const float*)d_in[20], *vel_b2 = (const float*)d_in[21];

  int N = in_sizes[0] / HD;   // 50000
  int E = in_sizes[3];        // 800000

  char* ws = (char*)d_ws;
  _Float16* msg_w1p = (_Float16*)(ws + 0);        // 64*160 f16 = 20480 B
  _Float16* msg_w2p = (_Float16*)(ws + 20480);    // 64*64       =  8192 B
  _Float16* pos_w1p = (_Float16*)(ws + 28672);    //             =  8192 B
  _Float16* nf_w1p  = (_Float16*)(ws + 36864);    // 64*128      = 16384 B
  _Float16* nf_w2p  = (_Float16*)(ws + 53248);    //             =  8192 B
  _Float16* vel_w1p = (_Float16*)(ws + 61440);    //             =  8192 B
  float* msg_agg = (float*)(ws + 69632);          // N*64 f32
  float* pos_agg = msg_agg + (size_t)N * 64;      // N*3
  float* cnt     = pos_agg + (size_t)N * 3;       // N
  _Float16* feat16 = (_Float16*)((char*)(cnt + N)); // N*64 f16 (16B-aligned)

  // pack weight matrices into WMMA B-fragment layout (f16)
  pack_B_kernel<<<(64 * KP_MSG + 255) / 256, 256, 0, stream>>>(msg_w1, msg_w1p, 2 * HD + 3, KP_MSG);
  pack_B_kernel<<<(64 * 64 + 255) / 256, 256, 0, stream>>>(msg_w2, msg_w2p, 64, 64);
  pack_B_kernel<<<(64 * 64 + 255) / 256, 256, 0, stream>>>(pos_w1, pos_w1p, 64, 64);
  pack_B_kernel<<<(64 * KP_NF + 255) / 256, 256, 0, stream>>>(nf_w1, nf_w1p, KP_NF, KP_NF);
  pack_B_kernel<<<(64 * 64 + 255) / 256, 256, 0, stream>>>(nf_w2, nf_w2p, 64, 64);
  pack_B_kernel<<<(64 * 64 + 255) / 256, 256, 0, stream>>>(vel_w1, vel_w1p, 64, 64);

  // node features -> f16 once (halves all gather traffic, kills cvt VALU)
  cvt_feat_kernel<<<(N * HD + 255) / 256, 256, 0, stream>>>(node_feat, feat16, N * HD);

  // zero accumulators (msg_agg | pos_agg | cnt are contiguous: N*68 floats)
  int zc = N * 68;
  zero_kernel<<<(zc + 255) / 256, 256, 0, stream>>>(msg_agg, zc);

  int etiles = (E + 31) / 32;
  egnn_edge_kernel<<<(etiles + WPB_E - 1) / WPB_E, 32 * WPB_E, 0, stream>>>(
      feat16, node_pos, edge_row, edge_col, edge_attr,
      msg_w1p, msg_b1, msg_w2p, msg_b2, pos_w1p, pos_b1, pos_w2, pos_b2,
      msg_agg, pos_agg, cnt, E);

  float* out_feat = (float*)d_out;
  float* out_pos  = out_feat + (size_t)N * 64;
  int ntiles = (N + 15) / 16;
  egnn_node_kernel<<<(ntiles + WPB_N - 1) / WPB_N, 32 * WPB_N, 0, stream>>>(
      feat16, node_pos, node_vel,
      nf_w1p, nf_b1, nf_w2p, nf_b2, vel_w1p, vel_b1, vel_w2, vel_b2,
      msg_agg, pos_agg, cnt, out_feat, out_pos, N);
}
